// CenterVLAD_79302276153947
// MI455X (gfx1250) — matrix operand order repaired and checked by reading, not compiled, not run on hardware
//
#include <hip/hip_runtime.h>
#include <stdint.h>

#define NIMG 32
#define HW   4096
#define DCH  512
#define KCL  116
#define KP   128
#define EPSV 1e-12f

typedef __attribute__((ext_vector_type(16))) __bf16 v16bf;
typedef __attribute__((ext_vector_type(8)))  float  v8f;
typedef __attribute__((ext_vector_type(4)))  float  f32x4;
typedef __attribute__((ext_vector_type(4)))  unsigned int tdm_u4;
typedef __attribute__((ext_vector_type(8)))  int          tdm_i8;
typedef __attribute__((ext_vector_type(4)))  int          tdm_i4;

union FragBF { v16bf v; uint4 q[2]; };

__device__ __forceinline__ unsigned short f2bf(float f) {
  unsigned u = __float_as_uint(f);
  u += 0x7FFFu + ((u >> 16) & 1u);         // round-to-nearest-even
  return (unsigned short)(u >> 16);
}

// Low 32 bits of a generic LDS pointer are the 0-based LDS byte offset.
__device__ __forceinline__ unsigned ldsOff(const void* p) {
  return (unsigned)(unsigned long long)(uintptr_t)p;
}

// TDM: DMA a 2D bf16 tile (128 rows x 64 cols, global row stride 512 elems)
// from global memory into LDS. Descriptor per CDNA5 ISA section 8.3/8.4.
__device__ __forceinline__ void tdm_load_wb_tile(const unsigned short* gsrc,
                                                 unsigned ldsAddr) {
#if __has_builtin(__builtin_amdgcn_tensor_load_to_lds)
  unsigned long long ga = (unsigned long long)(uintptr_t)gsrc;
  tdm_u4 g0;
  g0[0] = 1u;                                               // count=1, user desc
  g0[1] = ldsAddr;                                          // lds_addr (bytes)
  g0[2] = (unsigned)ga;                                     // global_addr[31:0]
  g0[3] = ((unsigned)(ga >> 32) & 0x01FFFFFFu) | 0x80000000u; // addr[56:32], type=2
  tdm_i8 g1;
  g1[0] = 0x10000;                  // workgroup_mask=0, data_size=1 (2 bytes)
  g1[1] = (int)(512u << 16);        // tensor_dim0 = 512 (bits 63:48)
  g1[2] = (int)(128u << 16);        // tensor_dim0_hi=0, tensor_dim1 = 128
  g1[3] = (int)(64u << 16);         // tensor_dim1_hi=0, tile_dim0 = 64
  g1[4] = 128;                      // tile_dim1 = 128, tile_dim2 = 0
  g1[5] = 512;                      // tensor_dim0_stride = 512 (low 32)
  g1[6] = 0;                        // stride hi / dim1_stride lo
  g1[7] = 0;
  tdm_i4 gz4 = {0, 0, 0, 0};
  tdm_i8 gz8 = {0, 0, 0, 0, 0, 0, 0, 0};
  __builtin_amdgcn_tensor_load_to_lds(g0, g1, gz4, gz4, gz8, 0);
#else
  (void)gsrc; (void)ldsAddr;
#endif
}

__device__ __forceinline__ void tdm_wait0() {
#if __has_builtin(__builtin_amdgcn_s_wait_tensorcnt)
  __builtin_amdgcn_s_wait_tensorcnt(0);
#endif
}

// ---------------- k0: zero accumulators, build wbT[k][d] (bf16) ----------------
__global__ __launch_bounds__(256) void k0_init(const float* __restrict__ conv_w,
                                               unsigned short* __restrict__ wbT,
                                               float* __restrict__ ax,
                                               float* __restrict__ asum) {
  unsigned i = blockIdx.x * 256u + threadIdx.x;
  if (i < (unsigned)(NIMG * DCH * KP)) ax[i] = 0.f;
  if (i < (unsigned)(KP * DCH)) {
    unsigned k = i >> 9, d = i & 511;
    wbT[i] = (k < KCL) ? f2bf(conv_w[d * KCL + k]) : (unsigned short)0;
  }
  if (i < (unsigned)(NIMG * KP)) asum[i] = 0.f;
}

// ------- k1: logits GEMM (WMMA bf16) + masked softmax + aT / asum writes -------
// TDM double-buffers the weight tile; x is staged (f32->bf16) by VALU.
__global__ __launch_bounds__(256) void k1_logits_softmax(
    const float* __restrict__ x, const unsigned short* __restrict__ wbT,
    unsigned short* __restrict__ aT, float* __restrict__ asum) {
  __shared__ __align__(16) unsigned short xs[128 * 64];     // 16 KB: 128 px x 64 ch
  __shared__ __align__(16) unsigned short wb[2][128 * 64];  // 2 x 16 KB ping-pong
  const int tid = threadIdx.x;
  const int n = blockIdx.x >> 5;
  const int pB = (blockIdx.x & 31) * 128;
  const int lane = tid & 31, wid = tid >> 5;
  const int r = lane & 15, hi = lane >> 4;

  v8f acc[8];
#pragma unroll
  for (int t = 0; t < 8; ++t) acc[t] = {};

  if (wid == 0) tdm_load_wb_tile(wbT, ldsOff(&wb[0][0]));   // prefetch slice 0

  const int prow = tid >> 1, dhalf = (tid & 1) * 32;
  for (int it = 0; it < 8; ++it) {
    const int d0 = it * 64;
    { // stage x slice as bf16 (non-temporal: x never fits L2, don't thrash aT)
      const f32x4* src = (const f32x4*)(x + ((size_t)(n * HW + pB + prow) * DCH + d0 + dhalf));
      unsigned short* dst = &xs[prow * 64 + dhalf];
#pragma unroll
      for (int i = 0; i < 8; ++i) {
        f32x4 f = __builtin_nontemporal_load(src + i);
        uint2 p;
        p.x = (unsigned)f2bf(f.x) | ((unsigned)f2bf(f.y) << 16);
        p.y = (unsigned)f2bf(f.z) | ((unsigned)f2bf(f.w) << 16);
        *(uint2*)(dst + i * 4) = p;
      }
    }
    if (wid == 0) tdm_wait0();          // wb[it&1] DMA complete
    __syncthreads();
    if (wid == 0 && it < 7)             // overlap next weight tile with compute
      tdm_load_wb_tile(wbT + d0 + 64, ldsOff(&wb[(it + 1) & 1][0]));
    const unsigned short* wbc = wb[it & 1];
#pragma unroll
    for (int kk = 0; kk < 2; ++kk) {
      FragBF A;
      const unsigned short* ap = &xs[(wid * 16 + r) * 64 + kk * 32 + hi * 8];
      A.q[0] = *(const uint4*)ap;
      A.q[1] = *(const uint4*)(ap + 16);
      FragBF B[8];
#pragma unroll
      for (int nt = 0; nt < 8; ++nt) {  // batch fragment loads: let DS pipeline
        const unsigned short* bp = &wbc[(nt * 16 + r) * 64 + kk * 32 + hi * 8];
        B[nt].q[0] = *(const uint4*)bp;
        B[nt].q[1] = *(const uint4*)(bp + 16);
      }
#pragma unroll
      for (int nt = 0; nt < 8; ++nt)    // back-to-back WMMA issue
        acc[nt] = __builtin_amdgcn_wmma_f32_16x16x32_bf16(false, A.v, false, B[nt].v,
                                                          (short)0, acc[nt], false, false);
    }
    __syncthreads();
  }

  // softmax over K (116 valid of 128); row values striped over 16 lanes x 8 tiles
#pragma unroll
  for (int j = 0; j < 8; ++j) {
    float vals[8];
    float m = -3.0e38f;
#pragma unroll
    for (int nt = 0; nt < 8; ++nt) {
      bool valid = (nt < 7) | (r < 4);
      vals[nt] = valid ? acc[nt][j] : -3.0e38f;
      m = fmaxf(m, vals[nt]);
    }
    m = fmaxf(m, __shfl_xor(m, 1));
    m = fmaxf(m, __shfl_xor(m, 2));
    m = fmaxf(m, __shfl_xor(m, 4));
    m = fmaxf(m, __shfl_xor(m, 8));
    float s = 0.f;
#pragma unroll
    for (int nt = 0; nt < 8; ++nt) {
      bool valid = (nt < 7) | (r < 4);
      float e = valid ? __expf(vals[nt] - m) : 0.f;
      vals[nt] = e;
      s += e;
    }
    s += __shfl_xor(s, 1);
    s += __shfl_xor(s, 2);
    s += __shfl_xor(s, 4);
    s += __shfl_xor(s, 8);
    float inv = 1.f / s;
#pragma unroll
    for (int nt = 0; nt < 8; ++nt) acc[nt][j] = vals[nt] * inv;
  }

  // write aT[n][k][p] (bf16, packed 16B) + asum atomics
#pragma unroll
  for (int nt = 0; nt < 8; ++nt) {
    float colsum = 0.f;
    unsigned short tmp[8];
#pragma unroll
    for (int j = 0; j < 8; ++j) {
      float a = acc[nt][j];
      colsum += a;
      tmp[j] = f2bf(a);
    }
    uint4 q;
    q.x = (unsigned)tmp[0] | ((unsigned)tmp[1] << 16);
    q.y = (unsigned)tmp[2] | ((unsigned)tmp[3] << 16);
    q.z = (unsigned)tmp[4] | ((unsigned)tmp[5] << 16);
    q.w = (unsigned)tmp[6] | ((unsigned)tmp[7] << 16);
    const int k = nt * 16 + r;
    atomicAdd(&asum[n * KP + k], colsum);
    *(uint4*)&aT[(size_t)(n * KP + k) * HW + pB + wid * 16 + hi * 8] = q;
  }
}

// ------------- k2: ax[n] += x^T * a  (WMMA bf16, split-K atomics) -------------
__global__ __launch_bounds__(256) void k2_ax(
    const float* __restrict__ x, const unsigned short* __restrict__ aT,
    float* __restrict__ ax) {
  __shared__ __align__(16) unsigned short xT[128 * 32];  // 8 KB: 128 d x 32 p (bf16)
  const int tid = threadIdx.x;
  const int n = blockIdx.x >> 4;
  const int dBase = ((blockIdx.x >> 2) & 3) * 128;
  const int pBase = (blockIdx.x & 3) * 1024;
  const int lane = tid & 31, wid = tid >> 5;
  const int r = lane & 15, hi = lane >> 4;

  v8f acc[8];
#pragma unroll
  for (int t = 0; t < 8; ++t) acc[t] = {};

  const int lp = tid >> 3, ldq = (tid & 7) * 16;
  for (int it = 0; it < 32; ++it) {
    const int p0 = pBase + it * 32;
    { // transpose-stage 32px x 128ch of x into xT[d][p] bf16
      const f32x4* src = (const f32x4*)(x + ((size_t)(n * HW + p0 + lp) * DCH + dBase + ldq));
#pragma unroll
      for (int i = 0; i < 4; ++i) {
        f32x4 f = __builtin_nontemporal_load(src + i);
        xT[(ldq + i * 4 + 0) * 32 + lp] = f2bf(f.x);
        xT[(ldq + i * 4 + 1) * 32 + lp] = f2bf(f.y);
        xT[(ldq + i * 4 + 2) * 32 + lp] = f2bf(f.z);
        xT[(ldq + i * 4 + 3) * 32 + lp] = f2bf(f.w);
      }
    }
    __syncthreads();
    FragBF A;
    const unsigned short* apt = &xT[(wid * 16 + r) * 32 + hi * 8];
    A.q[0] = *(const uint4*)apt;
    A.q[1] = *(const uint4*)(apt + 16);
    FragBF B[8];
#pragma unroll
    for (int nt = 0; nt < 8; ++nt) {    // batch global B-fragment loads (L2-hot aT)
      const unsigned short* bp = &aT[(size_t)(n * KP + nt * 16 + r) * HW + p0 + hi * 8];
      B[nt].q[0] = *(const uint4*)bp;
      B[nt].q[1] = *(const uint4*)(bp + 16);
    }
#pragma unroll
    for (int nt = 0; nt < 8; ++nt)
      acc[nt] = __builtin_amdgcn_wmma_f32_16x16x32_bf16(false, A.v, false, B[nt].v,
                                                        (short)0, acc[nt], false, false);
    __syncthreads();
  }
#pragma unroll
  for (int nt = 0; nt < 8; ++nt)
#pragma unroll
    for (int j = 0; j < 8; ++j) {
      const int d = dBase + wid * 16 + hi * 8 + j;
      const int k = nt * 16 + r;
      atomicAdd(&ax[((size_t)n * DCH + d) * KP + k], acc[nt][j]);
    }
}

// --------- k3a: per-(n,k) column sum-of-squares of v = ax + C*asum ----------
__global__ __launch_bounds__(256) void k3a_colnorm(const float* __restrict__ ax,
                                                   const float* __restrict__ C,
                                                   const float* __restrict__ asum,
                                                   float* __restrict__ tcol) {
  __shared__ float red[256];
  const int n = blockIdx.x / KCL, k = blockIdx.x % KCL;
  const int tid = threadIdx.x;
  const float cs = asum[n * KP + k];
  float s = 0.f;
  for (int d = tid; d < DCH; d += 256) {
    float v = ax[((size_t)n * DCH + d) * KP + k] + C[d * KCL + k] * cs;
    s += v * v;
  }
  red[tid] = s;
  __syncthreads();
  for (int off = 128; off; off >>= 1) {
    if (tid < off) red[tid] += red[tid + off];
    __syncthreads();
  }
  if (!tid) tcol[n * KP + k] = red[0];
}

// ---------------- k3b: two-level normalization + final output ----------------
__global__ __launch_bounds__(256) void k3b_out(const float* __restrict__ ax,
                                               const float* __restrict__ C,
                                               const float* __restrict__ asum,
                                               const float* __restrict__ tcol,
                                               float* __restrict__ out) {
  __shared__ float red[256];
  __shared__ float invc[KCL];
  __shared__ float csh[KCL];
  const int n = blockIdx.x, tid = threadIdx.x;
  float val = 0.f;
  if (tid < KCL) {
    float S = tcol[n * KP + tid];
    val = S / (S + EPSV);
    invc[tid] = rsqrtf(S + EPSV);
    csh[tid] = asum[n * KP + tid];
  }
  red[tid] = val;
  __syncthreads();
  for (int off = 128; off; off >>= 1) {
    if (tid < off) red[tid] += red[tid + off];
    __syncthreads();
  }
  const float scale = rsqrtf(red[0] + EPSV);
  const int total = DCH * KCL;
  for (int i = tid; i < total; i += 256) {
    const int d = i / KCL, k = i - d * KCL;
    float v = ax[((size_t)n * DCH + d) * KP + k] + C[i] * csh[k];
    out[(size_t)n * total + i] = v * invc[k] * scale;
  }
}

extern "C" void kernel_launch(void* const* d_in, const int* in_sizes, int n_in,
                              void* d_out, int out_size, void* d_ws, size_t ws_size,
                              hipStream_t stream) {
  (void)in_sizes; (void)n_in; (void)out_size; (void)ws_size;
  const float* x      = (const float*)d_in[0];
  const float* conv_w = (const float*)d_in[1];
  const float* C      = (const float*)d_in[2];
  float* out = (float*)d_out;

  char* ws = (char*)d_ws;
  unsigned short* wbT = (unsigned short*)(ws);            // 131072 B  (128k x 512d bf16)
  float* asum = (float*)(ws + 131072);                    // 16384 B   (32 x 128 f32)
  float* tcol = (float*)(ws + 147456);                    // 16384 B   (32 x 128 f32)
  float* ax   = (float*)(ws + 163840);                    // 8 MB      (32 x 512 x 128 f32)
  unsigned short* aT = (unsigned short*)(ws + 8552448);   // 32 MB     (32 x 128 x 4096 bf16)

  k0_init<<<8192, 256, 0, stream>>>(conv_w, wbT, ax, asum);
  k1_logits_softmax<<<NIMG * 32, 256, 0, stream>>>(x, wbT, aT, asum);
  k2_ax<<<NIMG * 16, 256, 0, stream>>>(x, aT, ax);
  k3a_colnorm<<<NIMG * KCL, 256, 0, stream>>>(ax, C, asum, tcol);
  k3b_out<<<NIMG, 256, 0, stream>>>(ax, C, asum, tcol, out);
}
